// RL_Valu_39444979646862
// MI455X (gfx1250) — matrix operand order repaired and checked by reading, not compiled, not run on hardware
//
#include <hip/hip_runtime.h>
#include <hip/hip_bf16.h>

typedef __attribute__((ext_vector_type(16))) __bf16 v16bf;
typedef __attribute__((ext_vector_type(8)))  float  v8f;
typedef __attribute__((ext_vector_type(4)))  int    v4i;

union Frag16 { v16bf v; uint4 q[2]; };

#define NN     50000
#define DD     256
#define HH     2048
#define AA     512
#define NNORM  4096
#define KK     32
#define MTILES 3125   // NN/16
#define ATILES 32     // AA/16
#define JTILES (HH / 16)

// ---- feature detection for CDNA5 async/TDM paths (compile-safe guards) ----
#if defined(__has_builtin)
# if __has_builtin(__builtin_amdgcn_tensor_load_to_lds)
#  define HAVE_TDM 1
# endif
# if __has_builtin(__builtin_amdgcn_global_load_async_to_lds_b128)
#  define HAVE_ASYNC 1
# endif
# if __has_builtin(__builtin_amdgcn_s_wait_tensorcnt)
#  define HAVE_WAIT_TENSOR 1
# endif
# if __has_builtin(__builtin_amdgcn_s_wait_asynccnt)
#  define HAVE_WAIT_ASYNC 1
# endif
#endif

static __device__ __forceinline__ v8f wmma_bf16(v16bf a, v16bf b, v8f c) {
    return __builtin_amdgcn_wmma_f32_16x16x32_bf16(false, a, false, b, (short)0, c, false, false);
}

static __device__ __forceinline__ void wait_tensor0() {
#if defined(HAVE_WAIT_TENSOR)
    __builtin_amdgcn_s_wait_tensorcnt(0);
#else
    asm volatile("s_wait_tensorcnt 0" ::: "memory");
#endif
}

static __device__ __forceinline__ void wait_async0() {
#if defined(HAVE_WAIT_ASYNC)
    __builtin_amdgcn_s_wait_asynccnt(0);
#else
    asm volatile("s_wait_asynccnt 0" ::: "memory");
#endif
}

#if defined(HAVE_TDM)
typedef __attribute__((ext_vector_type(4))) unsigned int u32x4_t;
typedef __attribute__((ext_vector_type(4))) int          i32x4_t;
typedef __attribute__((ext_vector_type(8))) int          i32x8_t;

// TDM D# per CDNA5 ISA: 2D tile, 16 rows x 256 bf16 (8KB),
// row stride 256 elements, data_size=2B, no multicast/pad/iterate.
static __device__ __forceinline__ void tdm_load_w1_tile(const __bf16* w1t, int tile,
                                                        __bf16* ldsdst) {
    unsigned long long ga = (unsigned long long)(uintptr_t)(w1t + (size_t)tile * 16 * DD);
    unsigned lds_addr = (unsigned)(uintptr_t)ldsdst;   // low 32 bits = LDS offset
    u32x4_t g0;
    g0.x = 1u;                                          // count=1, user mode
    g0.y = lds_addr;                                    // lds_addr [63:32]
    g0.z = (unsigned)(ga & 0xFFFFFFFFu);                // global_addr [95:64]
    g0.w = (unsigned)((ga >> 32) & 0x01FFFFFFu)         // global_addr [120:96]
         | 0x80000000u;                                 // type=2 ("image") [127:126]
    i32x8_t g1;
    g1[0] = (int)(1u << 16);        // workgroup_mask=0, data_size=1 (2 bytes)
    g1[1] = (int)(256u << 16);      // tensor_dim0=256 -> low16 at bits 63:48
    g1[2] = (int)(2048u << 16);     // dim0 hi16=0; tensor_dim1=2048 low16 at 95:80
    g1[3] = (int)(256u << 16);      // dim1 hi16=0; tile_dim0=256 at 127:112
    g1[4] = 16;                     // tile_dim1=16, tile_dim2=0
    g1[5] = 256;                    // tensor_dim0_stride low32 = 256
    g1[6] = 0;                      // stride hi, dim1_stride low
    g1[7] = 0;
    i32x4_t gz = {0, 0, 0, 0};
#if __clang_major__ >= 23
    i32x8_t gz8 = {0, 0, 0, 0, 0, 0, 0, 0};
    __builtin_amdgcn_tensor_load_to_lds(g0, g1, gz, gz, gz8, 0);
#else
    __builtin_amdgcn_tensor_load_to_lds(g0, g1, gz, gz, 0);
#endif
}
#endif // HAVE_TDM

#if defined(HAVE_ASYNC)
// per-thread 32B async copy: 256 threads move one 8KB tile
static __device__ __forceinline__ void async_copy_tile(const __bf16* w1t, int tile,
                                                       __bf16* ldsdst, int tid) {
    const char* gsrc = (const char*)(w1t + (size_t)tile * 16 * DD) + tid * 32;
    char*       ldst = (char*)ldsdst + tid * 32;
    __builtin_amdgcn_global_load_async_to_lds_b128(
        (__attribute__((address_space(1))) v4i*)(uintptr_t)gsrc,
        (__attribute__((address_space(3))) v4i*)(uintptr_t)ldst, 0, 0);
    __builtin_amdgcn_global_load_async_to_lds_b128(
        (__attribute__((address_space(1))) v4i*)(uintptr_t)(gsrc + 16),
        (__attribute__((address_space(3))) v4i*)(uintptr_t)(ldst + 16), 0, 0);
}
#endif // HAVE_ASYNC

// ---------------- Kernel 1: row norms + bf16 copies ----------------
__global__ __launch_bounds__(256) void norm_kernel(const float* __restrict__ feat,
                                                   __bf16* __restrict__ featbf,
                                                   __bf16* __restrict__ nbf,
                                                   float* __restrict__ rnorm) {
    int i = blockIdx.x;
    int t = threadIdx.x;
    float v = feat[(size_t)i * DD + t];
    __shared__ float red[256];
    red[t] = v * v;
    __syncthreads();
    for (int s = 128; s > 0; s >>= 1) {
        if (t < s) red[t] += red[t + s];
        __syncthreads();
    }
    float inv = 1.0f / fmaxf(sqrtf(red[0]), 1e-8f);
    featbf[(size_t)i * DD + t] = (__bf16)v;
    nbf[(size_t)i * DD + t]    = (__bf16)(v * inv);
    if (t == 0) rnorm[i] = inv;
}

// ---------------- Kernel 2: W1 -> bf16 transposed (K-major) ----------------
__global__ __launch_bounds__(256) void w1t_kernel(const float* __restrict__ W1,
                                                  __bf16* __restrict__ w1t) {
    int idx = blockIdx.x * 256 + threadIdx.x;   // h*256 + d
    int h = idx >> 8, d = idx & 255;
    w1t[idx] = (__bf16)W1[(size_t)d * HH + h];
}

// ---------------- Kernel 3: fused MLP via bf16 WMMA + LDS-staged B ----------------
__global__ __launch_bounds__(256) void mlp_kernel(const __bf16* __restrict__ featbf,
                                                  const __bf16* __restrict__ w1t,
                                                  const float* __restrict__ b1,
                                                  const float* __restrict__ W2,
                                                  const float* __restrict__ b2,
                                                  float* __restrict__ out_pred) {
    __shared__ __bf16 lds_b[2][16 * DD];        // double-buffered 8KB W1^T tiles
    int wave = threadIdx.x >> 5, lane = threadIdx.x & 31;
    int mtile = blockIdx.x * 8 + wave;
    bool active = (mtile < MTILES);
    int mt = active ? mtile : (MTILES - 1);     // clamp; keep all waves in barriers
    int m0 = mt * 16;

    const __bf16* arow = featbf + (size_t)(m0 + (lane & 15)) * DD;
    int koff = (lane >> 4) * 8;
    Frag16 Afr[8];
#pragma unroll
    for (int s = 0; s < 8; ++s) {
        int base = s * 32 + koff;
        Afr[s].q[0] = *(const uint4*)(arow + base);
        Afr[s].q[1] = *(const uint4*)(arow + base + 16);
    }

    float psum[8] = {0, 0, 0, 0, 0, 0, 0, 0};

#if defined(HAVE_TDM)
    if (wave == 0) tdm_load_w1_tile(w1t, 0, &lds_b[0][0]);
#elif defined(HAVE_ASYNC)
    async_copy_tile(w1t, 0, &lds_b[0][0], threadIdx.x);
#endif

    for (int jt = 0; jt < JTILES; ++jt) {
        int cur = jt & 1;
#if defined(HAVE_TDM)
        if (wave == 0) wait_tensor0();
        __syncthreads();                         // tile jt resident for all waves
        if (jt + 1 < JTILES && wave == 0)
            tdm_load_w1_tile(w1t, jt + 1, &lds_b[1 - cur][0]);
#elif defined(HAVE_ASYNC)
        wait_async0();
        __syncthreads();
        if (jt + 1 < JTILES)
            async_copy_tile(w1t, jt + 1, &lds_b[1 - cur][0], threadIdx.x);
#else
        cur = 0;                                 // sync fallback, single buffer
        {
            const uint4* gs = (const uint4*)(w1t + (size_t)jt * 16 * DD);
            uint4*       ls = (uint4*)&lds_b[0][0];
            ls[threadIdx.x]       = gs[threadIdx.x];
            ls[threadIdx.x + 256] = gs[threadIdx.x + 256];
        }
        __syncthreads();
#endif
        const __bf16* brow = &lds_b[cur][(lane & 15) * DD + (lane >> 4) * 8];
        v8f acc = {0, 0, 0, 0, 0, 0, 0, 0};
#pragma unroll
        for (int s = 0; s < 8; ++s) {
            Frag16 B;
            B.q[0] = *(const uint4*)(brow + s * 32);
            B.q[1] = *(const uint4*)(brow + s * 32 + 16);
            acc = wmma_bf16(Afr[s].v, B.v, acc);
        }
        int col = jt * 16 + (lane & 15);
        float b1v = b1[col], w2v = W2[col];
#pragma unroll
        for (int r = 0; r < 8; ++r) {
            float h = acc[r] + b1v;
            h = h > 0.0f ? h : 0.0f;
            psum[r] += h * w2v;
        }
#if !defined(HAVE_TDM) && !defined(HAVE_ASYNC)
        __syncthreads();                         // protect single buffer
#endif
    }

#pragma unroll
    for (int r = 0; r < 8; ++r)
        for (int off = 1; off < 16; off <<= 1)
            psum[r] += __shfl_xor(psum[r], off, 32);
    if (active && (lane & 15) == 0) {
        int mb = m0 + ((lane >> 4) << 3);
        float bias2 = b2[0];
#pragma unroll
        for (int r = 0; r < 8; ++r) out_pred[mb + r] = psum[r] + bias2;
    }
}

// ---------------- Kernel 4: sims GEMM + streaming top-33 ----------------
__global__ __launch_bounds__(256) void sims_topk_kernel(const __bf16* __restrict__ nbf,
                                                        const int* __restrict__ alist,
                                                        float* __restrict__ nn_sim_ws,
                                                        int* __restrict__ nn_idx_ws,
                                                        float* __restrict__ outf) {
    __shared__ float lval[8][16][33];
    __shared__ int   lidx[8][16][33];
    __shared__ float sval[256];
    __shared__ int   sidx[256];
    int wave = threadIdx.x >> 5, lane = threadIdx.x & 31;

    for (int i = threadIdx.x; i < 8 * 16 * 33; i += 256) {
        (&lval[0][0][0])[i] = -1e30f;
        (&lidx[0][0][0])[i] = 0;
    }
    __syncthreads();

    int aidx = alist[blockIdx.x * 16 + (lane & 15)];
    const __bf16* arow = nbf + (size_t)aidx * DD;
    int koff = (lane >> 4) * 8;
    Frag16 Afr[8];
#pragma unroll
    for (int s = 0; s < 8; ++s) {
        int base = s * 32 + koff;
        Afr[s].q[0] = *(const uint4*)(arow + base);
        Afr[s].q[1] = *(const uint4*)(arow + base + 16);
    }

    for (int ct = wave; ct < MTILES; ct += 8) {
        int node = ct * 16 + (lane & 15);
        const __bf16* brow = nbf + (size_t)node * DD + (lane >> 4) * 8;
        if (ct + 8 < MTILES)
            __builtin_prefetch(nbf + (size_t)(node + 128) * DD, 0, 3);
        v8f acc = {0, 0, 0, 0, 0, 0, 0, 0};
#pragma unroll
        for (int s = 0; s < 8; ++s) {
            Frag16 B;
            B.q[0] = *(const uint4*)(brow + s * 32);
            B.q[1] = *(const uint4*)(brow + s * 32 + 16);
            acc = wmma_bf16(Afr[s].v, B.v, acc);
        }
#pragma unroll
        for (int r = 0; r < 8; ++r) {
            int row = r + ((lane >> 4) << 3);
            float val = acc[r];
            float minv = lval[wave][row][32];
            bool pass = val > minv;
            unsigned long long bal = __ballot(pass);
            unsigned grp = (lane < 16) ? (unsigned)(bal & 0xffffu)
                                       : (unsigned)((bal >> 16) & 0xffffu);
            if (grp) {
                sval[threadIdx.x] = val;
                sidx[threadIdx.x] = node;
                asm volatile("s_wait_dscnt 0" ::: "memory");
                if ((lane & 15) == 0) {
                    float* L  = &lval[wave][row][0];
                    int*   LI = &lidx[wave][row][0];
                    unsigned m = grp;
                    int base = threadIdx.x;
                    while (m) {
                        int b = __ffs(m) - 1;
                        m &= m - 1;
                        float v = sval[base + b];
                        int  id = sidx[base + b];
                        if (v > L[32]) {
                            int i = 32;
                            while (i > 0 && L[i - 1] < v) {
                                L[i] = L[i - 1]; LI[i] = LI[i - 1]; --i;
                            }
                            L[i] = v; LI[i] = id;
                        }
                    }
                }
                asm volatile("s_wait_dscnt 0" ::: "memory");
            }
        }
    }
    __syncthreads();

    if (threadIdx.x < 16) {
        int row = threadIdx.x;
        int arow_g = blockIdx.x * 16 + row;
        int heads[8] = {0, 0, 0, 0, 0, 0, 0, 0};
        for (int out = 0; out < 33; ++out) {
            float best = -2e30f; int bw = 0;
            for (int w = 0; w < 8; ++w)
                if (heads[w] < 33) {
                    float v = lval[w][row][heads[w]];
                    if (v > best) { best = v; bw = w; }
                }
            int id = lidx[bw][row][heads[bw]];
            heads[bw]++;
            if (out >= 1) {
                int j = out - 1;
                nn_sim_ws[arow_g * KK + j] = best;
                nn_idx_ws[arow_g * KK + j] = id;
                outf[16384 + arow_g * KK + j] = (float)id;
            }
        }
    }
}

// ---------------- Kernel 5: centroids ----------------
__global__ __launch_bounds__(256) void centroid_kernel(const float* __restrict__ feat,
                                                       const float* __restrict__ rnorm,
                                                       const int* __restrict__ alist,
                                                       const int* __restrict__ nlist,
                                                       float* __restrict__ abc,
                                                       float* __restrict__ noc) {
    int d = threadIdx.x;
    float s = 0.0f;
    for (int a = 0; a < AA; ++a) {
        int i = alist[a];
        s += feat[(size_t)i * DD + d] * rnorm[i];
    }
    abc[d] = s * (1.0f / (float)AA);
    float s2 = 0.0f;
    for (int a = 0; a < NNORM; ++a) {
        int i = nlist[a];
        s2 += feat[(size_t)i * DD + d] * rnorm[i];
    }
    noc[d] = s2 * (1.0f / (float)NNORM);
}

// ---------------- Kernel 6: conditions + RL episodes + mask/th ----------------
__global__ __launch_bounds__(32) void episode_kernel(const float* __restrict__ feat,
                                                     const float* __restrict__ rnorm,
                                                     const int* __restrict__ alist,
                                                     const float* __restrict__ abc,
                                                     const float* __restrict__ noc,
                                                     const float* __restrict__ nn_sim,
                                                     const int* __restrict__ nn_idx,
                                                     float* __restrict__ outf) {
    int a = blockIdx.x, lane = threadIdx.x;
    int aidx = alist[a];
    float rn = rnorm[aidx];
    float s_ab = 0.0f, s_no = 0.0f;
    for (int d = lane; d < DD; d += 32) {
        float v = feat[(size_t)aidx * DD + d] * rn;
        s_ab += v * abc[d];
        s_no += v * noc[d];
    }
    for (int off = 16; off > 0; off >>= 1) {
        s_ab += __shfl_xor(s_ab, off, 32);
        s_no += __shfl_xor(s_no, off, 32);
    }
    int nid = nn_idx[a * KK + lane];
    float rn2 = rnorm[nid];
    float n_ab = 0.0f, n_no = 0.0f;
    for (int d = 0; d < DD; ++d) {
        float v = feat[(size_t)nid * DD + d] * rn2;
        n_ab += v * abc[d];
        n_no += v * noc[d];
    }
    bool cab = (n_ab >= s_ab);
    bool cno = (n_no <= s_no);
    float sim = nn_sim[a * KK + lane];
    float th = 0.5f;
#pragma unroll
    for (int e = 0; e < 10; ++e) {
        bool keep = (sim >= th);
        bool rbit = keep ? cab : cno;
        unsigned long long bal = __ballot(rbit);
        int reward = __popcll(bal & 0xffffffffull);
        th = (reward >= 16) ? th - 0.02f : th + 0.02f;
    }
    if (th >= 1.0f) th = 0.999f;
    if (th <= 0.0f) th = 0.001f;
    outf[a * KK + lane] = (sim > th) ? 1.0f : 0.0f;
    if (lane == 0) outf[32768 + a] = th;
}

// ---------------- Launch ----------------
extern "C" void kernel_launch(void* const* d_in, const int* in_sizes, int n_in,
                              void* d_out, int out_size, void* d_ws, size_t ws_size,
                              hipStream_t stream) {
    const float* feat  = (const float*)d_in[0];
    const float* W1    = (const float*)d_in[1];
    const float* b1    = (const float*)d_in[2];
    const float* W2    = (const float*)d_in[3];
    const float* b2    = (const float*)d_in[4];
    const int*   alist = (const int*)d_in[5];
    const int*   nlist = (const int*)d_in[6];
    float* outf = (float*)d_out;

    char* w = (char*)d_ws;
    __bf16* featbf = (__bf16*)w;                       w += (size_t)NN * DD * 2;
    __bf16* nbf    = (__bf16*)w;                       w += (size_t)NN * DD * 2;
    __bf16* w1t    = (__bf16*)w;                       w += (size_t)HH * DD * 2;
    float*  rnorm  = (float*)w;                        w += (size_t)NN * 4;
    float*  abc    = (float*)w;                        w += 256 * 4;
    float*  noc    = (float*)w;                        w += 256 * 4;
    float*  nn_sim = (float*)w;                        w += (size_t)AA * KK * 4;
    int*    nn_idx = (int*)w;

    norm_kernel<<<NN, 256, 0, stream>>>(feat, featbf, nbf, rnorm);
    w1t_kernel<<<HH, 256, 0, stream>>>(W1, w1t);
    mlp_kernel<<<(MTILES + 7) / 8, 256, 0, stream>>>(featbf, w1t, b1, W2, b2,
                                                     outf + 33280);
    sims_topk_kernel<<<ATILES, 256, 0, stream>>>(nbf, alist, nn_sim, nn_idx, outf);
    centroid_kernel<<<1, 256, 0, stream>>>(feat, rnorm, alist, nlist, abc, noc);
    episode_kernel<<<AA, 32, 0, stream>>>(feat, rnorm, alist, abc, noc,
                                          nn_sim, nn_idx, outf);
}
